// GNN_QNetwork_76776835383553
// MI455X (gfx1250) — matrix operand order repaired and checked by reading, not compiled, not run on hardware
//
#include <hip/hip_runtime.h>
#include <hip/hip_bf16.h>
#include <stddef.h>

// ---------------- vector types ----------------
typedef __attribute__((ext_vector_type(16))) _Float16 v16h;
typedef __attribute__((ext_vector_type(8)))  _Float16 v8h;
typedef __attribute__((ext_vector_type(4)))  _Float16 v4h;
typedef __attribute__((ext_vector_type(8)))  float    v8f;

#define HID 64
#define FE  8

// ================= GCN plumbing (bandwidth-bound, VALU) =================

__global__ void k_fill(float* __restrict__ p, float v, int n) {
    int t = blockIdx.x * blockDim.x + threadIdx.x;
    if (t < n) p[t] = v;
}

__global__ void k_count_deg(const int* __restrict__ dst, float* __restrict__ deg, int E) {
    int t = blockIdx.x * blockDim.x + threadIdx.x;
    if (t < E) atomicAdd(&deg[dst[t]], 1.0f);
}

__global__ void k_rsqrt(float* __restrict__ deg, int n) {
    int t = blockIdx.x * blockDim.x + threadIdx.x;
    if (t < n) {
        float d = deg[t];
        deg[t] = (d > 0.0f) ? rsqrtf(d) : 0.0f;   // becomes dis
    }
}

// out[i][j] = sum_k X[i][k] * W[k][j],  j < 64
__global__ void k_gemm_node(const float* __restrict__ X, const float* __restrict__ W,
                            float* __restrict__ out, int N, int K) {
    int t = blockIdx.x * blockDim.x + threadIdx.x;
    if (t >= N * HID) return;
    int i = t >> 6, j = t & 63;
    const float* xr = X + (size_t)i * K;
    float s = 0.0f;
    for (int k = 0; k < K; ++k) s += xr[k] * W[k * HID + j];
    out[t] = s;
}

// self-loop term: agg[i][j] = xw[i][j] * dis[i]^2
__global__ void k_selfloop(const float* __restrict__ xw, const float* __restrict__ dis,
                           float* __restrict__ agg, int N) {
    int t = blockIdx.x * blockDim.x + threadIdx.x;
    if (t >= N * HID) return;
    float d = dis[t >> 6];
    agg[t] = xw[t] * d * d;
}

// agg[dst][j] += xw[src][j] * dis[src]*dis[dst]
__global__ void k_agg_edges(const float* __restrict__ xw, const int* __restrict__ src,
                            const int* __restrict__ dst, const float* __restrict__ dis,
                            float* __restrict__ agg, int E) {
    int t = blockIdx.x * blockDim.x + threadIdx.x;
    if (t >= E * HID) return;
    int e = t >> 6, j = t & 63;
    int s = src[e], d = dst[e];
    float v = xw[(size_t)s * HID + j] * dis[s] * dis[d];
    atomicAdd(&agg[(size_t)d * HID + j], v);
}

__global__ void k_bias_relu(const float* __restrict__ agg, const float* __restrict__ b,
                            float* __restrict__ h, int N) {
    int t = blockIdx.x * blockDim.x + threadIdx.x;
    if (t >= N * HID) return;
    h[t] = fmaxf(agg[t] + b[t & 63], 0.0f);
}

__global__ void k_cvt_f16(const float* __restrict__ h, _Float16* __restrict__ o, int n) {
    int t = blockIdx.x * blockDim.x + threadIdx.x;
    if (t < n) o[t] = (_Float16)h[t];
}

// ================= Edge MLP on WMMA f16 (the 82-GFLOP hot loop) =================
// z = [h2[src] | h2[dst] | ea] (136 -> pad 160);  L1: 136->128 relu; L2: 128->64 relu; L3: 64->1.
// Per wave: one 16-edge M-tile. A/B fragments built per ISA 7.12.2 f16 layouts:
//   A elem e (lane l): m = l%16, k = 16*(e/8) + 8*(l/16) + (e%8)  -> two 16B contiguous LDS loads
//   B elem e (lane l): n = l%16, k = 16*(e/8) + 8*(l/16) + (e%8)                -> two 16B contiguous LDS loads
// Weights pre-staged into LDS in fragment order once per block.

#define WB1_H 20480   // 8 n-tiles * 5 k-chunks * 32 lanes * 16 halves
#define WB2_H 8192    // 4 * 4 * 32 * 16
// LDS byte layout (dynamic):
//   wb1[20480]h @0 | wb2[8192]h @40960 | b1[128]f @57344 | b2[64]f @57856 |
//   w3[64]f @58112 | z (4w*2560h) @58368 | z2 (4w*2048h) @78848 | z3 (4w*1024f) @95232
#define SMEM_BYTES 111616

__device__ inline v8f vzero8() { v8f r = {0.f,0.f,0.f,0.f,0.f,0.f,0.f,0.f}; return r; }

__global__ void __launch_bounds__(128)
k_edge_mlp(const _Float16* __restrict__ h2h,
           const int* __restrict__ esrc, const int* __restrict__ edst,
           const float* __restrict__ ea,
           const float* __restrict__ mb1, const float* __restrict__ mb2,
           const float* __restrict__ mW1, const float* __restrict__ mW2,
           const float* __restrict__ mW3, const float* __restrict__ mb3,
           float* __restrict__ q, int E) {
    extern __shared__ __align__(16) char smem_raw[];
    _Float16* wb1 = (_Float16*)smem_raw;
    _Float16* wb2 = wb1 + WB1_H;
    float*    b1l = (float*)(wb1 + (WB1_H + WB2_H));
    float*    b2l = b1l + 128;
    float*    w3l = b2l + 64;
    _Float16* zb  = (_Float16*)(w3l + 64);
    _Float16* z2b = zb + 4 * 2560;
    float*    z3b = (float*)(z2b + 4 * 2048);

    // ---- stage weights/biases into fragment-ordered LDS (once per block) ----
    for (int idx = threadIdx.x; idx < WB1_H; idx += blockDim.x) {
        int e2 = idx & 15, l = (idx >> 4) & 31, tc = idx >> 9;   // tc in [0,40)
        int c = tc % 5, t = tc / 5;
        int k = c * 32 + (e2 >> 3) * 16 + (l >> 4) * 8 + (e2 & 7);     // [0,160)
        int n = t * 16 + (l & 15);               // [0,128)
        wb1[idx] = (k < 136) ? (_Float16)mW1[k * 128 + n] : (_Float16)0.0f;
    }
    for (int idx = threadIdx.x; idx < WB2_H; idx += blockDim.x) {
        int e2 = idx & 15, l = (idx >> 4) & 31, tc = idx >> 9;   // tc in [0,16)
        int c = tc & 3, t = tc >> 2;
        int k = c * 32 + (e2 >> 3) * 16 + (l >> 4) * 8 + (e2 & 7);     // [0,128)
        int n = t * 16 + (l & 15);               // [0,64)
        wb2[idx] = (_Float16)mW2[k * 64 + n];
    }
    for (int i = threadIdx.x; i < 128; i += blockDim.x) b1l[i] = mb1[i];
    for (int i = threadIdx.x; i < 64;  i += blockDim.x) { b2l[i] = mb2[i]; w3l[i] = mW3[i]; }
    __syncthreads();

    const int lane = threadIdx.x & 31;
    const int wv   = threadIdx.x >> 5;
    const int l16  = lane & 15;
    const int lh   = lane >> 4;
    const float bias3 = mb3[0];

    _Float16* z  = zb  + wv * 2560;   // 16 rows x 160 halves
    _Float16* z2 = z2b + wv * 2048;   // 16 x 128
    float*    z3 = z3b + wv * 1024;   // 16 x 64

    const int nTiles = (E + 15) >> 4;
    const int per    = gridDim.x * 4;
    const int iters  = (nTiles + per - 1) / per;   // uniform across block

    for (int it = 0; it < iters; ++it) {
        int tile = it * per + blockIdx.x * 4 + wv;
        int e0 = tile * 16;

        // ---- stage z (16 x 160 f16): h2[src] | h2[dst] | ea | zero-pad ----
        #pragma unroll
        for (int r2 = 0; r2 < 8; ++r2) {
            int row = r2 * 2 + lh;
            int e = e0 + row; if (e >= E) e = E - 1; if (e < 0) e = 0;
            int s = esrc[e], d = edst[e];
            *(v4h*)(z + row * 160 + l16 * 4)      = *(const v4h*)(h2h + (size_t)s * HID + l16 * 4);
            *(v4h*)(z + row * 160 + 64 + l16 * 4) = *(const v4h*)(h2h + (size_t)d * HID + l16 * 4);
            #pragma unroll
            for (int c = l16; c < 32; c += 16) {
                int col = 128 + c;
                z[row * 160 + col] = (col < 136) ? (_Float16)ea[(size_t)e * FE + (col - 128)]
                                                 : (_Float16)0.0f;
            }
        }
        __syncthreads();

        // ---- layer 1: 16x160 @ 160x128, 8 n-tiles x 5 k-chunks ----
        v8f acc[8];
        #pragma unroll
        for (int t = 0; t < 8; ++t) acc[t] = vzero8();
        #pragma unroll
        for (int c = 0; c < 5; ++c) {
            v8h lo = *(const v8h*)(z + l16 * 160 + c * 32 + lh * 8);
            v8h hi = *(const v8h*)(z + l16 * 160 + c * 32 + lh * 8 + 16);
            v16h a = __builtin_shufflevector(lo, hi, 0,1,2,3,4,5,6,7,8,9,10,11,12,13,14,15);
            #pragma unroll
            for (int t = 0; t < 8; ++t) {
                const _Float16* wp = wb1 + ((t * 5 + c) * 32 + lane) * 16;
                v8h blo = *(const v8h*)(wp);
                v8h bhi = *(const v8h*)(wp + 8);
                v16h b = __builtin_shufflevector(blo, bhi, 0,1,2,3,4,5,6,7,8,9,10,11,12,13,14,15);
                acc[t] = __builtin_amdgcn_wmma_f32_16x16x32_f16(
                             false, a, false, b, (short)0, acc[t], false, false);
            }
        }
        // bias + relu -> z2 (f16, A-major 16x128)
        #pragma unroll
        for (int t = 0; t < 8; ++t) {
            int n = t * 16 + l16;
            float bb = b1l[n];
            #pragma unroll
            for (int v = 0; v < 8; ++v) {
                float xv = fmaxf(acc[t][v] + bb, 0.0f);
                z2[(lh * 8 + v) * 128 + n] = (_Float16)xv;
            }
        }
        __syncthreads();

        // ---- layer 2: 16x128 @ 128x64, 4 n-tiles x 4 k-chunks ----
        v8f acc2[4];
        #pragma unroll
        for (int t = 0; t < 4; ++t) acc2[t] = vzero8();
        #pragma unroll
        for (int c = 0; c < 4; ++c) {
            v8h lo = *(const v8h*)(z2 + l16 * 128 + c * 32 + lh * 8);
            v8h hi = *(const v8h*)(z2 + l16 * 128 + c * 32 + lh * 8 + 16);
            v16h a = __builtin_shufflevector(lo, hi, 0,1,2,3,4,5,6,7,8,9,10,11,12,13,14,15);
            #pragma unroll
            for (int t = 0; t < 4; ++t) {
                const _Float16* wp = wb2 + ((t * 4 + c) * 32 + lane) * 16;
                v8h blo = *(const v8h*)(wp);
                v8h bhi = *(const v8h*)(wp + 8);
                v16h b = __builtin_shufflevector(blo, bhi, 0,1,2,3,4,5,6,7,8,9,10,11,12,13,14,15);
                acc2[t] = __builtin_amdgcn_wmma_f32_16x16x32_f16(
                              false, a, false, b, (short)0, acc2[t], false, false);
            }
        }
        // bias + relu -> z3 (f32 16x64)
        #pragma unroll
        for (int t = 0; t < 4; ++t) {
            int n = t * 16 + l16;
            float bb = b2l[n];
            #pragma unroll
            for (int v = 0; v < 8; ++v)
                z3[(lh * 8 + v) * 64 + n] = fmaxf(acc2[t][v] + bb, 0.0f);
        }
        __syncthreads();

        // ---- layer 3: per-edge 64-dot (split across lane halves, shfl reduce) ----
        float p = 0.0f;
        #pragma unroll
        for (int j = 0; j < 32; ++j)
            p += z3[l16 * 64 + lh * 32 + j] * w3l[lh * 32 + j];
        p += __shfl_down(p, 16, 32);
        int eo = e0 + l16;
        if (lh == 0 && tile < nTiles && eo < E) q[eo] = p + bias3;
    }
}

// ================= host-side launch =================

static inline size_t alignup(size_t x) { return (x + 255) & ~(size_t)255; }

extern "C" void kernel_launch(void* const* d_in, const int* in_sizes, int n_in,
                              void* d_out, int out_size, void* d_ws, size_t ws_size,
                              hipStream_t stream) {
    const float* x    = (const float*)d_in[0];
    const int*   ei   = (const int*)  d_in[1];
    const float* ea   = (const float*)d_in[2];
    const float* W1   = (const float*)d_in[3];
    const float* b1   = (const float*)d_in[4];
    const float* W2   = (const float*)d_in[5];
    const float* b2   = (const float*)d_in[6];
    const float* mW1  = (const float*)d_in[7];
    const float* mb1  = (const float*)d_in[8];
    const float* mW2  = (const float*)d_in[9];
    const float* mb2  = (const float*)d_in[10];
    const float* mW3  = (const float*)d_in[11];
    const float* mb3  = (const float*)d_in[12];
    float* q = (float*)d_out;

    const int N = in_sizes[0] / 16;       // nodes
    const int E = in_sizes[1] / 2;        // edges
    const int K1 = in_sizes[3] / HID;     // = 16
    const int* esrc = ei;
    const int* edst = ei + E;

    // workspace carve
    char* w = (char*)d_ws;
    size_t o = 0;
    float* dis = (float*)(w + o); o += alignup((size_t)N * 4);
    float* bufA = (float*)(w + o); o += alignup((size_t)N * HID * 4);
    float* bufB = (float*)(w + o); o += alignup((size_t)N * HID * 4);
    float* bufC = (float*)(w + o); o += alignup((size_t)N * HID * 4);
    _Float16* h2h = (_Float16*)(w + o); o += alignup((size_t)N * HID * 2);
    (void)ws_size; (void)n_in; (void)out_size;

    const int T = 256;
    auto g = [&](long long n) { return (int)((n + T - 1) / T); };

    // degrees (self-loops start at 1) -> dis = rsqrt(deg)
    k_fill<<<g(N), T, 0, stream>>>(dis, 1.0f, N);
    k_count_deg<<<g(E), T, 0, stream>>>(edst, dis, E);
    k_rsqrt<<<g(N), T, 0, stream>>>(dis, N);

    // GCN layer 1
    k_gemm_node<<<g((long long)N * HID), T, 0, stream>>>(x, W1, bufA, N, K1);
    k_selfloop<<<g((long long)N * HID), T, 0, stream>>>(bufA, dis, bufB, N);
    k_agg_edges<<<g((long long)E * HID), T, 0, stream>>>(bufA, esrc, edst, dis, bufB, E);
    k_bias_relu<<<g((long long)N * HID), T, 0, stream>>>(bufB, b1, bufA, N);   // h1 -> A

    // GCN layer 2
    k_gemm_node<<<g((long long)N * HID), T, 0, stream>>>(bufA, W2, bufC, N, HID);
    k_selfloop<<<g((long long)N * HID), T, 0, stream>>>(bufC, dis, bufB, N);
    k_agg_edges<<<g((long long)E * HID), T, 0, stream>>>(bufC, esrc, edst, dis, bufB, E);
    k_bias_relu<<<g((long long)N * HID), T, 0, stream>>>(bufB, b2, bufC, N);   // h2 -> C
    k_cvt_f16<<<g((long long)N * HID), T, 0, stream>>>(bufC, h2h, N * HID);

    // Edge MLP (WMMA f16)
    int grid = 1024;
    k_edge_mlp<<<grid, 128, SMEM_BYTES, stream>>>(h2h, esrc, edst, ea,
                                                  mb1, mb2, mW1, mW2, mW3, mb3,
                                                  q, E);
}
